// BatchedSRNNCell_87660282511869
// MI455X (gfx1250) — matrix operand order repaired and checked by reading, not compiled, not run on hardware
//
#include <hip/hip_runtime.h>
#include <hip/hip_bf16.h>

// Problem constants (match reference)
#define KNET 8
#define NN   512
#define NE_  256
#define MA_  3
#define IN_  128
#define BB   1024
#define BT   16
#define STATE_ 2560              // NE*MA + NI*MA + NE + NI + N
#define UNFOLDS_ 6
#define DT_  (0.04f / 6.0f)
#define SPAD 528                 // padded f16 stride for s in LDS (16B-aligned rows)
#define TPB  512                 // 16 waves; each wave owns 32 neuron rows (2 M-tiles)

typedef __attribute__((ext_vector_type(16))) _Float16 v16h;
typedef __attribute__((ext_vector_type(8)))  float    v8f;

union Frag16 { v16h v; uint4 u[2]; };

__device__ __forceinline__ float sp_(float x) { return log1pf(__expf(x)); }

// piecewise_sigmoid with S_a=0.9, S_c=0 -> a=0.45, k=5, x1=-0.55, x2=-0.45, x3=0.45, x4=0.55
__device__ __forceinline__ float psig(float x) {
  float t1 = x + 0.55f;
  float qr = 5.0f * t1 * t1;
  float lin = x + 0.5f;
  float t4 = x - 0.55f;
  float qc = 1.0f - 5.0f * t4 * t4;
  float o = (x < -0.55f) ? 0.0f : qr;
  o = (x >= -0.45f) ? lin : o;
  o = (x >   0.45f) ? qc  : o;
  o = (x >   0.55f) ? 1.0f : o;
  return o;
}

// ---- prep: W_eff = [softplus(W)[:, :NE], -softplus(W)[:, NE:]] * mask  -> f16 row-major ----
__global__ __launch_bounds__(256) void prep_weff(const float* __restrict__ wraw,
                                                 const float* __restrict__ mask,
                                                 _Float16* __restrict__ weff) {
  int i = blockIdx.x * 256 + threadIdx.x;
  if (i >= KNET * NN * NN) return;
  float w = log1pf(__expf(wraw[i]));
  float v = w * mask[i];
  if ((i & (NN - 1)) >= NE_) v = -v;   // presynaptic column >= NE -> inhibitory
  weff[i] = (_Float16)v;
}

// ---- prep: f32 -> f16 conversions for W_in and inputs ----
__global__ __launch_bounds__(256) void prep_f16(const float* __restrict__ win,
                                                const float* __restrict__ inp,
                                                _Float16* __restrict__ winh,
                                                _Float16* __restrict__ inph) {
  int i = blockIdx.x * 256 + threadIdx.x;
  if (i < KNET * NN * IN_) winh[i] = (_Float16)win[i];
  if (i < BB * IN_)        inph[i] = (_Float16)inp[i];
}

// ---- fused persistent recurrence kernel: one WG per (k, 16-batch tile) ----
__global__ __launch_bounds__(TPB) void srnn_step_kernel(
    const float* __restrict__ state,
    const float* __restrict__ a0g,
    const float* __restrict__ ltd,
    const float* __restrict__ ltaE, const float* __restrict__ lcE, const float* __restrict__ c0E,
    const float* __restrict__ ltaI, const float* __restrict__ lcI, const float* __restrict__ c0I,
    const float* __restrict__ lbrecE, const float* __restrict__ lbrelE,
    const float* __restrict__ lbrecI, const float* __restrict__ lbrelI,
    const int*   __restrict__ rids,
    const _Float16* __restrict__ weff,
    const _Float16* __restrict__ winh,
    const _Float16* __restrict__ inph,
    float* __restrict__ out,
    float* __restrict__ nstate) {
  __shared__ __align__(16) _Float16 s_lds[BT * SPAD];   // s, column(batch)-major
  __shared__ float4 P0[NN], P1[NN], P2[NN];             // per-neuron coefficients

  const int k     = blockIdx.y;
  const int btile = blockIdx.x;
  const int tid   = threadIdx.x;
  const int lane  = tid & 31;
  const int wave  = tid >> 5;
  const int la    = lane & 15;   // A row / B column / D column within tile
  const int hi    = lane >> 4;   // lane-half (K-interleave select / D row offset)

  // ---- per-neuron coefficient precompute into LDS (one neuron per thread) ----
  {
    int n = tid;  // TPB == NN
    float gx = DT_ / sp_(ltd[k * NN + n]);
    float px = 1.0f / (1.0f + gx);
    float qx = gx * px;
    float grec, grel, thr0;
    float pa[3], ka[3];
    if (n < NE_) {
      int e = k * NE_ + n;
      grec = DT_ / sp_(lbrecE[e]);
      grel = DT_ / sp_(lbrelE[e]);
      thr0 = a0g[k * NN + n];
#pragma unroll
      for (int m = 0; m < MA_; ++m) {
        thr0 += c0E[e * MA_ + m];
        float ga = DT_ / sp_(ltaE[e * MA_ + m]);
        float c  = sp_(lcE[e * MA_ + m]);
        pa[m] = 1.0f / (1.0f + ga);
        ka[m] = ga * c * pa[m];
      }
    } else {
      int e = k * NE_ + (n - NE_);
      grec = DT_ / sp_(lbrecI[e]);
      grel = DT_ / sp_(lbrelI[e]);
      thr0 = a0g[k * NN + n];
#pragma unroll
      for (int m = 0; m < MA_; ++m) {
        thr0 += c0I[e * MA_ + m];
        float ga = DT_ / sp_(ltaI[e * MA_ + m]);
        float c  = sp_(lcI[e * MA_ + m]);
        pa[m] = 1.0f / (1.0f + ga);
        ka[m] = ga * c * pa[m];
      }
    }
    P0[n] = make_float4(px, qx, grec, grel);
    P1[n] = make_float4(pa[0], pa[1], pa[2], thr0);
    P2[n] = make_float4(ka[0], ka[1], ka[2], 0.0f);
  }
  __syncthreads();

  // ---- load state into registers, laid out to match the WMMA D matrix ----
  const int b = btile * BT + la;                 // batch column owned by this lane
  const size_t kb = (size_t)k * BB + b;
  const float* stp = state + kb * STATE_;
  const int rb = wave * 32;                      // this wave's row base (2 tiles of 16)

  float xs[2][8], as[2][8][3], bs[2][8];
#pragma unroll
  for (int t = 0; t < 2; ++t)
#pragma unroll
    for (int j = 0; j < 8; ++j) {
      int n = rb + t * 16 + j + hi * 8;
      xs[t][j] = stp[2048 + n];
      if (n < NE_) {
#pragma unroll
        for (int m = 0; m < 3; ++m) as[t][j][m] = stp[n * 3 + m];
        bs[t][j] = stp[1536 + n];
      } else {
        int ni = n - NE_;
#pragma unroll
        for (int m = 0; m < 3; ++m) as[t][j][m] = stp[768 + ni * 3 + m];
        bs[t][j] = stp[1792 + ni];
      }
    }

  // ---- in_cur = W_in[k] x inputs^T for this batch tile (WMMA, K=128) ----
  v8f incur[2];
  {
    v8f c0 = {0.f, 0.f, 0.f, 0.f, 0.f, 0.f, 0.f, 0.f};
    v8f c1 = c0;
    const _Float16* ar0 = winh + ((size_t)k * NN + rb + la) * IN_ + hi * 8;
    const _Float16* ar1 = ar0 + (size_t)16 * IN_;
    const _Float16* bp  = inph + (size_t)b * IN_ + hi * 8;
#pragma unroll
    for (int ks = 0; ks < 4; ++ks) {
      int m0 = ks * 32;
      Frag16 fb, fa0, fa1;
      fb.u[0]  = *reinterpret_cast<const uint4*>(bp + m0);
      fb.u[1]  = *reinterpret_cast<const uint4*>(bp + m0 + 16);
      fa0.u[0] = *reinterpret_cast<const uint4*>(ar0 + m0);
      fa0.u[1] = *reinterpret_cast<const uint4*>(ar0 + m0 + 16);
      fa1.u[0] = *reinterpret_cast<const uint4*>(ar1 + m0);
      fa1.u[1] = *reinterpret_cast<const uint4*>(ar1 + m0 + 16);
      c0 = __builtin_amdgcn_wmma_f32_16x16x32_f16(false, fa0.v, false, fb.v, (short)0, c0,
                                                  false, false);
      c1 = __builtin_amdgcn_wmma_f32_16x16x32_f16(false, fa1.v, false, fb.v, (short)0, c1,
                                                  false, false);
    }
    incur[0] = c0;
    incur[1] = c1;
  }

  // ---- 6-step recurrence ----
  const _Float16* wk = weff + (size_t)k * NN * NN;
#pragma unroll 1
  for (int step = 0; step < UNFOLDS_; ++step) {
    // phase 1: s = r * b from current state -> LDS (f16, B-matrix column layout)
#pragma unroll
    for (int t = 0; t < 2; ++t)
#pragma unroll
      for (int j = 0; j < 8; ++j) {
        int n = rb + t * 16 + j + hi * 8;
        float4 p1 = P1[n];
        float thr = p1.w + as[t][j][0] + as[t][j][1] + as[t][j][2];
        float r = psig(xs[t][j] - thr);
        s_lds[la * SPAD + n] = (_Float16)(r * bs[t][j]);
      }
    __syncthreads();

    // phase 2: syn = W_eff x s + in_cur  (f16 WMMA, f32 accumulate)
    v8f acc0 = incur[0];
    v8f acc1 = incur[1];
    {
      const _Float16* ar0 = wk + (size_t)(rb + la) * NN + hi * 8;
      const _Float16* ar1 = ar0 + (size_t)16 * NN;
      const _Float16* bp  = &s_lds[la * SPAD + hi * 8];
#pragma unroll 4
      for (int ks = 0; ks < 16; ++ks) {
        int m0 = ks * 32;
        Frag16 fb, fa0, fa1;
        fb.u[0]  = *reinterpret_cast<const uint4*>(bp + m0);
        fb.u[1]  = *reinterpret_cast<const uint4*>(bp + m0 + 16);
        fa0.u[0] = *reinterpret_cast<const uint4*>(ar0 + m0);
        fa0.u[1] = *reinterpret_cast<const uint4*>(ar0 + m0 + 16);
        fa1.u[0] = *reinterpret_cast<const uint4*>(ar1 + m0);
        fa1.u[1] = *reinterpret_cast<const uint4*>(ar1 + m0 + 16);
        acc0 = __builtin_amdgcn_wmma_f32_16x16x32_f16(false, fa0.v, false, fb.v, (short)0,
                                                      acc0, false, false);
        acc1 = __builtin_amdgcn_wmma_f32_16x16x32_f16(false, fa1.v, false, fb.v, (short)0,
                                                      acc1, false, false);
      }
    }
    __syncthreads();

    // phase 3: elementwise state update (r recomputed from still-old state)
#pragma unroll
    for (int t = 0; t < 2; ++t)
#pragma unroll
      for (int j = 0; j < 8; ++j) {
        int n = rb + t * 16 + j + hi * 8;
        float4 p0 = P0[n];
        float4 p1 = P1[n];
        float4 p2 = P2[n];
        float thr = p1.w + as[t][j][0] + as[t][j][1] + as[t][j][2];
        float r = psig(xs[t][j] - thr);
        float syn = (t == 0) ? acc0[j] : acc1[j];
        xs[t][j] = p0.x * xs[t][j] + p0.y * syn;
        as[t][j][0] = p1.x * as[t][j][0] + p2.x * r;
        as[t][j][1] = p1.y * as[t][j][1] + p2.y * r;
        as[t][j][2] = p1.z * as[t][j][2] + p2.z * r;
        float den = 1.0f + p0.z + p0.w * r;
        bs[t][j] = (bs[t][j] + p0.z) * __builtin_amdgcn_rcpf(den);
      }
  }

  // ---- final rate/syn + readout select + state writeback ----
  int rid = rids[k];
  float* op  = out    + kb * NN;
  float* nsp = nstate + kb * STATE_;
#pragma unroll
  for (int t = 0; t < 2; ++t)
#pragma unroll
    for (int j = 0; j < 8; ++j) {
      int n = rb + t * 16 + j + hi * 8;
      float4 p1 = P1[n];
      float thr = p1.w + as[t][j][0] + as[t][j][1] + as[t][j][2];
      float r = psig(xs[t][j] - thr);
      float s = r * bs[t][j];
      float ov = (rid == 0) ? s : ((rid == 1) ? r : xs[t][j]);
      op[n] = ov;
      nsp[2048 + n] = xs[t][j];
      if (n < NE_) {
#pragma unroll
        for (int m = 0; m < 3; ++m) nsp[n * 3 + m] = as[t][j][m];
        nsp[1536 + n] = bs[t][j];
      } else {
        int ni = n - NE_;
#pragma unroll
        for (int m = 0; m < 3; ++m) nsp[768 + ni * 3 + m] = as[t][j][m];
        nsp[1792 + ni] = bs[t][j];
      }
    }
}

extern "C" void kernel_launch(void* const* d_in, const int* in_sizes, int n_in,
                              void* d_out, int out_size, void* d_ws, size_t ws_size,
                              hipStream_t stream) {
  const float* inputs = (const float*)d_in[0];
  const float* state  = (const float*)d_in[1];
  const float* wraw   = (const float*)d_in[2];
  const float* mask   = (const float*)d_in[3];
  const float* win    = (const float*)d_in[4];
  const float* a0g    = (const float*)d_in[5];
  const float* ltd    = (const float*)d_in[6];
  const float* ltaE   = (const float*)d_in[7];
  const float* lcE    = (const float*)d_in[8];
  const float* c0E    = (const float*)d_in[9];
  const float* ltaI   = (const float*)d_in[10];
  const float* lcI    = (const float*)d_in[11];
  const float* c0I    = (const float*)d_in[12];
  const float* lbrecE = (const float*)d_in[13];
  const float* lbrelE = (const float*)d_in[14];
  const float* lbrecI = (const float*)d_in[15];
  const float* lbrelI = (const float*)d_in[16];
  const int*   rids   = (const int*)d_in[17];

  // workspace: f16 W_eff (4 MB) | f16 W_in (1 MB) | f16 inputs (256 KB)
  _Float16* weff = (_Float16*)d_ws;
  _Float16* winh = weff + (size_t)KNET * NN * NN;
  _Float16* inph = winh + (size_t)KNET * NN * IN_;

  float* out    = (float*)d_out;                       // (K,B,N)
  float* nstate = out + (size_t)KNET * BB * NN;        // (K,B,STATE)

  prep_weff<<<(KNET * NN * NN + 255) / 256, 256, 0, stream>>>(wraw, mask, weff);
  prep_f16<<<(KNET * NN * IN_ + 255) / 256, 256, 0, stream>>>(win, inputs, winh, inph);

  dim3 grid(BB / BT, KNET);
  srnn_step_kernel<<<grid, TPB, 0, stream>>>(state, a0g, ltd, ltaE, lcE, c0E,
                                             ltaI, lcI, c0I, lbrecE, lbrelE, lbrecI, lbrelI,
                                             rids, weff, winh, inph, out, nstate);
}
